// AttentionLayer_79293686218841
// MI455X (gfx1250) — matrix operand order repaired and checked by reading, not compiled, not run on hardware
//
#include <hip/hip_runtime.h>

// ---------------------------------------------------------------------------
// Types for CDNA5 WMMA (wave32)
// ---------------------------------------------------------------------------
typedef __attribute__((ext_vector_type(8)))  __bf16 v8bf;
typedef __attribute__((ext_vector_type(16))) __bf16 v16bf;
typedef __attribute__((ext_vector_type(8)))  float  v8f;
typedef __attribute__((ext_vector_type(4)))  unsigned int v4u;
typedef __attribute__((ext_vector_type(8)))  int    v8i;
typedef __attribute__((ext_vector_type(4)))  int    v4i;

#define B_DIM   4
#define N_DIM   2048
#define DMODEL  512
#define H_DIM   8
#define DHEAD   64
#define BIG_NEG (-3.4028235e38f)

#if __has_builtin(__builtin_amdgcn_tensor_load_to_lds) && __has_builtin(__builtin_amdgcn_s_wait_tensorcnt)
#define HAVE_TDM 1
#else
#define HAVE_TDM 0
#endif

static __device__ __forceinline__ v16bf cat8(v8bf lo, v8bf hi) {
    return __builtin_shufflevector(lo, hi, 0,1,2,3,4,5,6,7,8,9,10,11,12,13,14,15);
}

// A-fragment (16x32 bf16, M x K): lane row = lane&15;
// elements 0..7 -> K = k0 + 8*laneHi ; elements 8..15 -> K = k0 + 16 + 8*laneHi
static __device__ __forceinline__ v16bf load_afrag(const __bf16* rowbase, int k0, int laneHi) {
    v8bf c0 = *(const v8bf*)(rowbase + k0 + 8 * laneHi);
    v8bf c1 = *(const v8bf*)(rowbase + k0 + 16 + 8 * laneHi);
    return cat8(c0, c1);
}

// B-fragment (32x16 bf16, K x N): lane col = lane&15;
// elements 0..7 -> K = k0 + 8*laneHi ; elements 8..15 -> K = k0 + 16 + 8*laneHi
static __device__ __forceinline__ v16bf load_bfrag(const __bf16* colbase, int k0, int laneHi) {
    v8bf c0 = *(const v8bf*)(colbase + k0 + 8 * laneHi);
    v8bf c1 = *(const v8bf*)(colbase + k0 + 16 + 8 * laneHi);
    return cat8(c0, c1);
}

static __device__ __forceinline__ v8f wmma_bf16(v16bf a, v16bf b, v8f c) {
    return __builtin_amdgcn_wmma_f32_16x16x32_bf16(false, a, false, b, (short)0, c, false, false);
}

// ---------------------------------------------------------------------------
// Tensor Data Mover: 2D tile (bf16) load Global -> LDS.  Builds D# per ISA §8:
//   group0: count=1 | lds_addr | global_addr[56:0] | type=2
//   group1: data_size=1(2B), tensor_dim0/1, tile_dim0/1, tensor_dim0_stride
// ---------------------------------------------------------------------------
#if HAVE_TDM
static __device__ __forceinline__ void tdm_load_2d(
    unsigned lds_addr, const void* gptr,
    unsigned tensor_d0, unsigned tensor_d1,
    unsigned tile_d0, unsigned tile_d1, unsigned stride0) {
    unsigned long long ga = (unsigned long long)(size_t)gptr;
    v4u g0;
    g0[0] = 1u;                                            // count=1 (valid user D#)
    g0[1] = lds_addr;                                      // lds_addr  (bits 63:32)
    g0[2] = (unsigned)(ga & 0xffffffffu);                  // global_addr lo
    g0[3] = (unsigned)((ga >> 32) & 0x01ffffffu) | (2u << 30); // addr hi | type=2
    v8i g1;
    g1[0] = (int)(1u << 16);                               // wg_mask=0, data_size=1 (2B)
    g1[1] = (int)((tensor_d0 & 0xffffu) << 16);            // tensor_dim0[15:0] @ bit48
    g1[2] = (int)((tensor_d0 >> 16) | ((tensor_d1 & 0xffffu) << 16));
    g1[3] = (int)((tensor_d1 >> 16) | (tile_d0 << 16));    // tile_dim0 @ bit112
    g1[4] = (int)tile_d1;                                  // tile_dim1, tile_dim2=0
    g1[5] = (int)stride0;                                  // tensor_dim0_stride lo
    g1[6] = 0;
    g1[7] = 0;
    v4i z4 = {0, 0, 0, 0};
#if defined(__clang_major__) && (__clang_major__ >= 23)
    v8i z8 = {0, 0, 0, 0, 0, 0, 0, 0};
    __builtin_amdgcn_tensor_load_to_lds(g0, g1, z4, z4, z8, 0);
#else
    __builtin_amdgcn_tensor_load_to_lds(g0, g1, z4, z4, 0);
#endif
}
#endif

// ---------------------------------------------------------------------------
// Kernel 1: fp32 -> bf16 copy of x
// ---------------------------------------------------------------------------
__global__ void cvt_x_kernel(const float* __restrict__ x, __bf16* __restrict__ xb, int n) {
    int i = blockIdx.x * blockDim.x + threadIdx.x;
    if (i < n) xb[i] = (__bf16)x[i];
}

// Kernel 2: weight transpose + convert: WT[n][k] = (bf16) W[k][n]  (512x512)
__global__ void transpose_w_kernel(const float* __restrict__ W, __bf16* __restrict__ WT) {
    int k = blockIdx.x * 16 + threadIdx.x;
    int n = blockIdx.y * 16 + threadIdx.y;
    WT[n * DMODEL + k] = (__bf16)W[k * DMODEL + n];
}

// ---------------------------------------------------------------------------
// Software-pipelined 16x64-per-wave K-loop (shared by both GEMM kernels).
// Loads step k+32 into fresh locals while step k's 4 WMMAs issue, so the
// scheduler can hoist loads above the WMMA pack (no s_wait_loadcnt 0 stalls).
// ---------------------------------------------------------------------------
static __device__ __forceinline__ void gemm_kloop(
    const __bf16* arow, const __bf16* brow0, const __bf16* brow1,
    const __bf16* brow2, const __bf16* brow3, int laneHi,
    v8f& c0, v8f& c1, v8f& c2, v8f& c3) {
    v16bf a  = load_afrag(arow, 0, laneHi);
    v16bf b0 = load_bfrag(brow0, 0, laneHi);
    v16bf b1 = load_bfrag(brow1, 0, laneHi);
    v16bf b2 = load_bfrag(brow2, 0, laneHi);
    v16bf b3 = load_bfrag(brow3, 0, laneHi);
    for (int k0 = 0; k0 < DMODEL - 32; k0 += 32) {
        v16bf an  = load_afrag(arow, k0 + 32, laneHi);
        v16bf bn0 = load_bfrag(brow0, k0 + 32, laneHi);
        v16bf bn1 = load_bfrag(brow1, k0 + 32, laneHi);
        v16bf bn2 = load_bfrag(brow2, k0 + 32, laneHi);
        v16bf bn3 = load_bfrag(brow3, k0 + 32, laneHi);
        c0 = wmma_bf16(a, b0, c0);
        c1 = wmma_bf16(a, b1, c1);
        c2 = wmma_bf16(a, b2, c2);
        c3 = wmma_bf16(a, b3, c3);
        a = an; b0 = bn0; b1 = bn1; b2 = bn2; b3 = bn3;
    }
    c0 = wmma_bf16(a, b0, c0);
    c1 = wmma_bf16(a, b1, c1);
    c2 = wmma_bf16(a, b2, c2);
    c3 = wmma_bf16(a, b3, c3);
}

// ---------------------------------------------------------------------------
// Kernel 3: projection GEMM, 16x64 per wave (A-frag reused over 4 B-frags).
//   mode 0: q/k layout [b,h,n,d]  (scale folded)
//   mode 2: v transposed layout [b,h,d,n]
// ---------------------------------------------------------------------------
__global__ void __launch_bounds__(256)
gemm_qkv_kernel(const __bf16* __restrict__ xb, const __bf16* __restrict__ WT,
                __bf16* __restrict__ out, int mode, float scale) {
    int wid    = (blockIdx.x * blockDim.x + threadIdx.x) >> 5;
    int lane   = threadIdx.x & 31;
    int laneHi = lane >> 4;
    int laneLo = lane & 15;
    int mt  = wid >> 3;              // 512 M-tiles
    int ntb = wid & 7;               // 8 N-tile groups of 64 cols
    int m0  = mt << 4, n0 = ntb << 6;

    const __bf16* arow  = xb + (size_t)(m0 + laneLo) * DMODEL;
    const __bf16* brow0 = WT + (size_t)(n0 + 0  + laneLo) * DMODEL;
    const __bf16* brow1 = WT + (size_t)(n0 + 16 + laneLo) * DMODEL;
    const __bf16* brow2 = WT + (size_t)(n0 + 32 + laneLo) * DMODEL;
    const __bf16* brow3 = WT + (size_t)(n0 + 48 + laneLo) * DMODEL;

    v8f c0 = {}, c1 = {}, c2 = {}, c3 = {};
    gemm_kloop(arow, brow0, brow1, brow2, brow3, laneHi, c0, c1, c2, c3);

    v8f acc[4] = {c0, c1, c2, c3};
#pragma unroll
    for (int t = 0; t < 4; ++t) {
        int col = n0 + t * 16 + laneLo;       // 0..511
        int h = col >> 6, d = col & 63;
#pragma unroll
        for (int r = 0; r < 8; ++r) {
            int row = m0 + r + 8 * laneHi;    // token 0..8191
            int bb  = row >> 11;
            int n   = row & (N_DIM - 1);
            int bh  = bb * H_DIM + h;
            float val = acc[t][r] * scale;
            size_t off;
            if (mode == 2) off = ((size_t)bh * DHEAD + d) * N_DIM + n;   // [b,h,d,n]
            else           off = ((size_t)bh * N_DIM + n) * DHEAD + d;   // [b,h,n,d]
            out[off] = (__bf16)val;
        }
    }
}

// ---------------------------------------------------------------------------
// Kernel 4: flash attention. 4 waves/block share one (b,h); K/V tiles staged
// in LDS via TDM (double-buffered, TENSORcnt-fenced) or cooperative copies.
// ---------------------------------------------------------------------------
__global__ void __launch_bounds__(128)
attn_kernel(const __bf16* __restrict__ qb, const __bf16* __restrict__ kb,
            const __bf16* __restrict__ vT,
            const unsigned char* __restrict__ mask_q,
            const unsigned char* __restrict__ mask_k,
            __bf16* __restrict__ ao) {
    __shared__ __align__(16) __bf16 Ktile[2][32 * DHEAD];   // [key][d]
    __shared__ __align__(16) __bf16 Vtile[2][DHEAD * 32];   // [d][key]
    __shared__ __align__(16) __bf16 Pbuf[4][16 * 32];

    int tid    = threadIdx.x;
    int wslot  = tid >> 5;
    int lane   = tid & 31;
    int laneHi = lane >> 4;
    int laneLo = lane & 15;

    int bh = blockIdx.x >> 5;                 // b*8+h (32 blocks per bh)
    int b  = bh >> 3, h = bh & 7;
    int q0 = (((blockIdx.x & 31) << 2) + wslot) << 4;

    const __bf16* Q = qb + ((size_t)bh * N_DIM + q0) * DHEAD;
    const __bf16* K = kb + (size_t)bh * N_DIM * DHEAD;
    const __bf16* V = vT + (size_t)bh * DHEAD * N_DIM;

    // Q A-fragments for d-chunks 0..31 / 32..63
    const __bf16* qrow = Q + laneLo * DHEAD;
    v16bf qa0 = load_afrag(qrow, 0, laneHi);
    v16bf qa1 = load_afrag(qrow, 32, laneHi);

    unsigned char mqr[8];
#pragma unroll
    for (int r = 0; r < 8; ++r)
        mqr[r] = mask_q[b * N_DIM + q0 + r + 8 * laneHi];

    float m[8], l[8];
#pragma unroll
    for (int r = 0; r < 8; ++r) { m[r] = -__builtin_inff(); l[r] = 0.0f; }
    v8f acc[4] = {v8f{}, v8f{}, v8f{}, v8f{}};

    __bf16* P = &Pbuf[wslot][0];

    // ---- tile stage helper (TDM from wave 0, or cooperative copy) ----------
    auto fill_tiles = [&](int buf, int j0) {
#if HAVE_TDM
        if (tid < 32) {
            // K tile: 32 keys x 64 d from [2048 x 64] tensor, row stride 64
            tdm_load_2d((unsigned)(size_t)&Ktile[buf][0], K + (size_t)j0 * DHEAD,
                        DHEAD, N_DIM, DHEAD, 32, DHEAD);
            // V tile: 64 d x 32 keys from [64 x 2048] tensor, row stride 2048
            tdm_load_2d((unsigned)(size_t)&Vtile[buf][0], V + j0,
                        N_DIM, DHEAD, 32, DHEAD, N_DIM);
        }
#else
        // K tile region is fully contiguous (rows j0..j0+31 of 64 bf16)
        const v8bf* ks = (const v8bf*)(K + (size_t)j0 * DHEAD);
        v8bf* kd = (v8bf*)&Ktile[buf][0];
        for (int i = tid; i < 256; i += 128) kd[i] = ks[i];
        {   // V: 64 rows x two 16B chunks
            int d = tid >> 1, cchunk = (tid & 1) * 8;
            *(v8bf*)(&Vtile[buf][0] + d * 32 + cchunk) =
                *(const v8bf*)(V + (size_t)d * N_DIM + j0 + cchunk);
        }
#endif
    };
    auto tiles_ready = [&]() {
#if HAVE_TDM
        if (tid < 32) __builtin_amdgcn_s_wait_tensorcnt(0);
#endif
        __syncthreads();
    };

    fill_tiles(0, 0);
    tiles_ready();

    for (int j0 = 0; j0 < N_DIM; j0 += 32) {
        int cur = (j0 >> 5) & 1;
        if (j0 + 32 < N_DIM) fill_tiles(1 - cur, j0 + 32);   // overlap DMA w/ compute
        const __bf16* Kt = &Ktile[cur][0];
        const __bf16* Vt = &Vtile[cur][0];

        // ---- S = Q K^T (scale folded into q), two 16-key sub-tiles
        v8f S[2];
#pragma unroll
        for (int sub = 0; sub < 2; ++sub) {
            const __bf16* krow = Kt + (sub * 16 + laneLo) * DHEAD;
            v16bf kb0 = load_bfrag(krow, 0, laneHi);
            v16bf kb1 = load_bfrag(krow, 32, laneHi);
            v8f s = {};
            s = wmma_bf16(qa0, kb0, s);
            s = wmma_bf16(qa1, kb1, s);
            unsigned char km = mask_k[b * N_DIM + j0 + sub * 16 + laneLo];
#pragma unroll
            for (int r = 0; r < 8; ++r)
                S[sub][r] = (km && mqr[r]) ? s[r] : BIG_NEG;
        }

        // ---- online softmax per row (rows live in one half-wave)
#pragma unroll
        for (int r = 0; r < 8; ++r) {
            float smax = fmaxf(S[0][r], S[1][r]);
#pragma unroll
            for (int off = 8; off; off >>= 1)
                smax = fmaxf(smax, __shfl_xor(smax, off, 16));
            float mnew = fmaxf(m[r], smax);
            float sc   = __expf(m[r] - mnew);
            float p0   = __expf(S[0][r] - mnew);
            float p1   = __expf(S[1][r] - mnew);
            float rs   = p0 + p1;
#pragma unroll
            for (int off = 8; off; off >>= 1)
                rs += __shfl_xor(rs, off, 16);
            l[r] = l[r] * sc + rs;
            m[r] = mnew;
            acc[0][r] *= sc; acc[1][r] *= sc; acc[2][r] *= sc; acc[3][r] *= sc;
            int prow = r + 8 * laneHi;
            P[prow * 32 + laneLo]      = (__bf16)p0;
            P[prow * 32 + 16 + laneLo] = (__bf16)p1;
        }

        // ---- P (D-layout in LDS) -> A-fragment, then O += P @ V
        const __bf16* prow = P + laneLo * 32;
        v16bf pa = load_afrag(prow, 0, laneHi);
#pragma unroll
        for (int t = 0; t < 4; ++t) {
            const __bf16* vrow = Vt + (t * 16 + laneLo) * 32;
            v16bf vb = load_bfrag(vrow, 0, laneHi);
            acc[t] = wmma_bf16(pa, vb, acc[t]);
        }

        tiles_ready();   // fence next tile (TENSORcnt + block barrier)
    }

    // ---- normalize, store ao[token][h*64+d] bf16
#pragma unroll
    for (int r = 0; r < 8; ++r) {
        float inv = 1.0f / l[r];
        size_t token = (size_t)b * N_DIM + q0 + r + 8 * laneHi;
#pragma unroll
        for (int t = 0; t < 4; ++t) {
            int d = t * 16 + laneLo;
            ao[token * (H_DIM * DHEAD) + h * DHEAD + d] = (__bf16)(acc[t][r] * inv);
        }
    }
}

// ---------------------------------------------------------------------------
// Kernel 5: output GEMM  out = ao[8192x512] @ Wo (via WoT) + bo, fp32 out
// ---------------------------------------------------------------------------
__global__ void __launch_bounds__(256)
gemm_out_kernel(const __bf16* __restrict__ ao, const __bf16* __restrict__ WoT,
                const float* __restrict__ bo, float* __restrict__ out) {
    int wid    = (blockIdx.x * blockDim.x + threadIdx.x) >> 5;
    int lane   = threadIdx.x & 31;
    int laneHi = lane >> 4;
    int laneLo = lane & 15;
    int mt  = wid >> 3;
    int ntb = wid & 7;
    int m0  = mt << 4, n0 = ntb << 6;

    const __bf16* arow  = ao  + (size_t)(m0 + laneLo) * DMODEL;
    const __bf16* brow0 = WoT + (size_t)(n0 + 0  + laneLo) * DMODEL;
    const __bf16* brow1 = WoT + (size_t)(n0 + 16 + laneLo) * DMODEL;
    const __bf16* brow2 = WoT + (size_t)(n0 + 32 + laneLo) * DMODEL;
    const __bf16* brow3 = WoT + (size_t)(n0 + 48 + laneLo) * DMODEL;

    v8f c0 = {}, c1 = {}, c2 = {}, c3 = {};
    gemm_kloop(arow, brow0, brow1, brow2, brow3, laneHi, c0, c1, c2, c3);

    v8f acc[4] = {c0, c1, c2, c3};
#pragma unroll
    for (int t = 0; t < 4; ++t) {
        int col  = n0 + t * 16 + laneLo;
        float bias = bo[col];
#pragma unroll
        for (int r = 0; r < 8; ++r) {
            int row = m0 + r + 8 * laneHi;
            out[(size_t)row * DMODEL + col] = acc[t][r] + bias;
        }
    }
}

// ---------------------------------------------------------------------------
// Host launcher
// ---------------------------------------------------------------------------
extern "C" void kernel_launch(void* const* d_in, const int* in_sizes, int n_in,
                              void* d_out, int out_size, void* d_ws, size_t ws_size,
                              hipStream_t stream) {
    (void)in_sizes; (void)n_in; (void)out_size; (void)ws_size;

    const float* x          = (const float*)d_in[0];
    const unsigned char* mk = (const unsigned char*)d_in[1];  // mask_k (bool)
    const unsigned char* mq = (const unsigned char*)d_in[2];  // mask_q (bool)
    const float* Wq         = (const float*)d_in[3];
    const float* Wk         = (const float*)d_in[4];
    const float* Wv         = (const float*)d_in[5];
    const float* Wo         = (const float*)d_in[6];
    const float* bo         = (const float*)d_in[7];
    float* outp             = (float*)d_out;

    const size_t TOK = (size_t)B_DIM * N_DIM;        // 8192
    const size_t XEL = TOK * DMODEL;                 // 4,194,304
    const size_t WEL = (size_t)DMODEL * DMODEL;      // 262,144

    char* w = (char*)d_ws;
    size_t off = 0;
    __bf16* xb  = (__bf16*)(w + off); off += XEL * 2;
    __bf16* WqT = (__bf16*)(w + off); off += WEL * 2;
    __bf16* WkT = (__bf16*)(w + off); off += WEL * 2;
    __bf16* WvT = (__bf16*)(w + off); off += WEL * 2;
    __bf16* WoT = (__bf16*)(w + off); off += WEL * 2;
    __bf16* qb  = (__bf16*)(w + off); off += XEL * 2;  // [b,h,n,d]
    __bf16* kbp = (__bf16*)(w + off); off += XEL * 2;  // [b,h,n,d]
    __bf16* vT  = (__bf16*)(w + off); off += XEL * 2;  // [b,h,d,n]
    __bf16* ao  = (__bf16*)(w + off); off += XEL * 2;  // [token, 512]

    // 1) convert x to bf16
    cvt_x_kernel<<<(unsigned)((XEL + 255) / 256), 256, 0, stream>>>(x, xb, (int)XEL);

    // 2) transpose+convert weights
    dim3 tb(16, 16), tg(DMODEL / 16, DMODEL / 16);
    transpose_w_kernel<<<tg, tb, 0, stream>>>(Wq, WqT);
    transpose_w_kernel<<<tg, tb, 0, stream>>>(Wk, WkT);
    transpose_w_kernel<<<tg, tb, 0, stream>>>(Wv, WvT);
    transpose_w_kernel<<<tg, tb, 0, stream>>>(Wo, WoT);

    // 3) q/k/v projections: 512 M-tiles x 8 N-groups = 4096 waves -> 512 blocks
    const unsigned gemmBlocks = 512;
    gemm_qkv_kernel<<<gemmBlocks, 256, 0, stream>>>(xb, WqT, qb, 0, 0.125f);
    gemm_qkv_kernel<<<gemmBlocks, 256, 0, stream>>>(xb, WkT, kbp, 0, 1.0f);
    gemm_qkv_kernel<<<gemmBlocks, 256, 0, stream>>>(xb, WvT, vT, 2, 1.0f);

    // 4) attention: 4096 waves, 4 waves/block sharing one (b,h)
    attn_kernel<<<1024, 128, 0, stream>>>(qb, kbp, vT, mq, mk, ao);

    // 5) output projection + bias (fp32 out)
    gemm_out_kernel<<<gemmBlocks, 256, 0, stream>>>(ao, WoT, bo, outp);
}
